// GATNet_30124900614318
// MI455X (gfx1250) — matrix operand order repaired and checked by reading, not compiled, not run on hardware
//
#include <hip/hip_runtime.h>
#include <math.h>

#define N_NODES 50000
#define N_EDGES 800000
#define D_IN    128
#define HEADS   8
#define HC      16
#define H_DIM   128
#define G_GR    64
#define OUT_C   16
#define EPS_BN  1e-5f

typedef __attribute__((ext_vector_type(16))) __bf16 v16bf;
typedef __attribute__((ext_vector_type(8)))  float  v8f;

#define CDIV(a, b) (((a) + (b) - 1) / (b))

// ---------------- utility ----------------

__global__ void fill_kernel(float* __restrict__ p, float v, int n) {
    int i = blockIdx.x * blockDim.x + threadIdx.x;
    if (i < n) p[i] = v;
}

__device__ inline void atomicMaxF(float* addr, float val) {
    // valid when destination initialized to -inf
    if (val >= 0.0f) atomicMax((int*)addr, __float_as_int(val));
    else             atomicMin((unsigned int*)addr, (unsigned int)__float_as_int(val));
}

__device__ inline float lrelu02(float v) { return v > 0.0f ? v : 0.2f * v; }

// ---------------- batch norm (eval) ----------------

__global__ void bn_kernel(const float* __restrict__ x,
                          const float* __restrict__ gamma, const float* __restrict__ beta,
                          const float* __restrict__ mean,  const float* __restrict__ var,
                          float* __restrict__ out, int total) {
    int i = blockIdx.x * blockDim.x + threadIdx.x;
    if (i >= total) return;
    int d = i & (D_IN - 1);
    float v = (x[i] - mean[d]) * rsqrtf(var[d] + EPS_BN) * gamma[d] + beta[d];
    out[i] = v;
}

// ---------------- graph constants: degree + mean incoming edge attr ----------------

__global__ void deg_easum_kernel(const int* __restrict__ dst, const float* __restrict__ ew,
                                 float* __restrict__ deg, float* __restrict__ easum, int e_cnt) {
    int i = blockIdx.x * blockDim.x + threadIdx.x;
    if (i >= e_cnt) return;
    int d = dst[i];
    atomicAdd(&deg[d], 1.0f);
    atomicAdd(&easum[d], ew[i]);
}

__global__ void eamean_kernel(const float* __restrict__ deg, float* __restrict__ ea, int n) {
    int i = blockIdx.x * blockDim.x + threadIdx.x;
    if (i >= n) return;
    ea[i] = ea[i] / fmaxf(deg[i], 1.0f);
}

// ---------------- ce[h] = sum_k We[h*16+k] * att_edge[h*16+k] ----------------

__global__ void ce_kernel(const float* __restrict__ We, const float* __restrict__ att_e,
                          float* __restrict__ ce) {
    int h = threadIdx.x;
    if (h >= HEADS) return;
    float s = 0.0f;
    for (int k = 0; k < HC; ++k) s += We[h * HC + k] * att_e[h * HC + k];
    ce[h] = s;
}

// ---------------- pre-pack W (f32 row-major) into bf16 WMMA B-operand layout ----------
// Wpk[((c*4 + kb)*32 + lane)*16 + e] = bf16( W[(kb*32 + e + 16*(lane>>4))*128 + c*16 + (lane&15)] )
// so each lane of the GEMM loads its 16 B elements as one contiguous 32-byte chunk.

__global__ void packW_kernel(const float* __restrict__ W, __bf16* __restrict__ Wpk) {
    int i = blockIdx.x * blockDim.x + threadIdx.x;      // 8*4*32*16 = 16384
    if (i >= 16384) return;
    int e    = i & 15;
    int lane = (i >> 4) & 31;
    int kb   = (i >> 9) & 3;
    int c    = i >> 11;
    int k = kb * 32 + e + 16 * (lane >> 4);
    int n = c * 16 + (lane & 15);
    Wpk[i] = (__bf16)W[k * H_DIM + n];
}

// ---------------- WMMA GEMM: C[N,128] = X[N,128] * W[128,128] (bf16 in, f32 acc) ------
// One wave computes a 16x128 output strip: 8 accumulators, A loaded once per k-block,
// B loaded as packed contiguous v16bf. 32 static v_wmma per wave.

__global__ void __launch_bounds__(128)
gemm_wmma_kernel(const float* __restrict__ X, const __bf16* __restrict__ Wpk,
                 float* __restrict__ C) {
    const int wave = threadIdx.x >> 5;
    const int lane = threadIdx.x & 31;
    const int strip = blockIdx.x * 4 + wave;             // 3125 strips (N/16)
    if (strip >= N_NODES / 16) return;                   // wave-uniform exit
    const int row0 = strip << 4;
    const int m  = lane & 15;
    const int hi = lane >> 4;

    v8f acc[8];
    #pragma unroll
    for (int c = 0; c < 8; ++c) acc[c] = (v8f){};

    const float* arow = X + (size_t)(row0 + m) * H_DIM;
    #pragma unroll
    for (int kb = 0; kb < 4; ++kb) {
        const int kbase = kb * 32;
        v16bf a;
        // A: 16x32 tile, lane holds row m; run [kbase+16*hi, +16)
        #pragma unroll
        for (int t = 0; t < 8; ++t) {
            a[t]     = (__bf16)arow[kbase + 16 * hi + t];
            a[t + 8] = (__bf16)arow[kbase + 16 * hi + 8 + t];
        }
        #pragma unroll
        for (int c = 0; c < 8; ++c) {
            const v16bf b = *reinterpret_cast<const v16bf*>(
                Wpk + (((size_t)(c * 4 + kb) * 32 + lane) << 4));
            acc[c] = __builtin_amdgcn_wmma_f32_16x16x32_bf16(false, a, false, b,
                                                             (short)0, acc[c], false, false);
        }
    }
    // C/D layout: VGPR r -> row r + 8*hi, col = c*16 + lane%16
    #pragma unroll
    for (int c = 0; c < 8; ++c) {
        #pragma unroll
        for (int r = 0; r < 8; ++r) {
            C[(size_t)(row0 + r + 8 * hi) * H_DIM + c * 16 + m] = acc[c][r];
        }
    }
}

// ---------------- per-node attention coefficients ----------------

__global__ void node_coef_kernel(const float* __restrict__ hlin,
                                 const float* __restrict__ att_s, const float* __restrict__ att_d,
                                 float* __restrict__ aS, float* __restrict__ aD, int total) {
    int i = blockIdx.x * blockDim.x + threadIdx.x;      // total = N*HEADS
    if (i >= total) return;
    int n = i >> 3, h = i & 7;
    const float* hp = hlin + (size_t)n * H_DIM + h * HC;
    float s1 = 0.0f, s2 = 0.0f;
    #pragma unroll
    for (int k = 0; k < HC; ++k) {
        float v = hp[k];
        s1 += v * att_s[h * HC + k];
        s2 += v * att_d[h * HC + k];
    }
    aS[i] = s1;
    aD[i] = s2;
}

// ---------------- edge passes (self-loops appended: e in [E, E+N)) ----------------

__device__ inline void edge_sdw(int e, const int* src, const int* dst,
                                const float* ew, const float* eamean,
                                int& s, int& d, float& w) {
    if (e < N_EDGES) { s = src[e]; d = dst[e]; w = ew[e]; }
    else             { s = e - N_EDGES; d = s; w = eamean[s]; }
}

__global__ void edge_max_kernel(const int* __restrict__ src, const int* __restrict__ dst,
                                const float* __restrict__ ew, const float* __restrict__ eamean,
                                const float* __restrict__ aS, const float* __restrict__ aD,
                                const float* __restrict__ ce, float* __restrict__ amax, int total) {
    int i = blockIdx.x * blockDim.x + threadIdx.x;      // total = (E+N)*HEADS
    if (i >= total) return;
    int e = i >> 3, h = i & 7;
    int s, d; float w;
    edge_sdw(e, src, dst, ew, eamean, s, d, w);
    float alpha = lrelu02(aS[s * 8 + h] + aD[d * 8 + h] + w * ce[h]);
    atomicMaxF(&amax[d * 8 + h], alpha);
}

__global__ void edge_expsum_kernel(const int* __restrict__ src, const int* __restrict__ dst,
                                   const float* __restrict__ ew, const float* __restrict__ eamean,
                                   const float* __restrict__ aS, const float* __restrict__ aD,
                                   const float* __restrict__ ce, const float* __restrict__ amax,
                                   float* __restrict__ denom, int total) {
    int i = blockIdx.x * blockDim.x + threadIdx.x;
    if (i >= total) return;
    int e = i >> 3, h = i & 7;
    int s, d; float w;
    edge_sdw(e, src, dst, ew, eamean, s, d, w);
    float alpha = lrelu02(aS[s * 8 + h] + aD[d * 8 + h] + w * ce[h]);
    atomicAdd(&denom[d * 8 + h], __expf(alpha - amax[d * 8 + h]));
}

__global__ void edge_scatter_kernel(const int* __restrict__ src, const int* __restrict__ dst,
                                    const float* __restrict__ ew, const float* __restrict__ eamean,
                                    const float* __restrict__ aS, const float* __restrict__ aD,
                                    const float* __restrict__ ce, const float* __restrict__ amax,
                                    const float* __restrict__ denom, const float* __restrict__ hlin,
                                    float* __restrict__ out, int total) {
    int i = blockIdx.x * blockDim.x + threadIdx.x;      // total = (E+N)*32 (4 comps per thread)
    if (i >= total) return;
    int e = i >> 5, c4 = i & 31;
    int h = c4 >> 2, comp = c4 << 2;
    int s, d; float w;
    edge_sdw(e, src, dst, ew, eamean, s, d, w);
    float alpha = lrelu02(aS[s * 8 + h] + aD[d * 8 + h] + w * ce[h]);
    float coef = __expf(alpha - amax[d * 8 + h]) / (denom[d * 8 + h] + 1e-16f);
    const float4 hv = *reinterpret_cast<const float4*>(hlin + (size_t)s * H_DIM + comp);
    float* op = out + (size_t)d * H_DIM + comp;
    atomicAdd(op + 0, hv.x * coef);
    atomicAdd(op + 1, hv.y * coef);
    atomicAdd(op + 2, hv.z * coef);
    atomicAdd(op + 3, hv.w * coef);
}

__global__ void bias_act_kernel(const float* __restrict__ acc, const float* __restrict__ b,
                                float* __restrict__ out, int total, int do_relu) {
    int i = blockIdx.x * blockDim.x + threadIdx.x;
    if (i >= total) return;
    float v = acc[i] + b[i & (H_DIM - 1)];
    out[i] = do_relu ? fmaxf(v, 0.0f) : v;
}

// ---------------- global max pool + MLP head ----------------

__global__ void pool_kernel(const float* __restrict__ hfin, const int* __restrict__ batch,
                            float* __restrict__ g, int total) {
    int i = blockIdx.x * blockDim.x + threadIdx.x;      // total = N*128
    if (i >= total) return;
    int n = i >> 7, c = i & 127;
    atomicMaxF(&g[(size_t)batch[n] * H_DIM + c], hfin[i]);
}

__global__ void mlp1_kernel(const float* __restrict__ g, const float* __restrict__ Wl1,
                            const float* __restrict__ bl1, float* __restrict__ t, int total) {
    int i = blockIdx.x * blockDim.x + threadIdx.x;      // total = G*32
    if (i >= total) return;
    int gr = i >> 5, j = i & 31;
    float s = bl1[j];
    for (int c = 0; c < H_DIM; ++c) s += g[gr * H_DIM + c] * Wl1[c * 32 + j];
    t[i] = fmaxf(s, 0.0f);
}

__global__ void mlp2_kernel(const float* __restrict__ t, const float* __restrict__ Wl2,
                            const float* __restrict__ bl2, float* __restrict__ out, int total) {
    int i = blockIdx.x * blockDim.x + threadIdx.x;      // total = G*16
    if (i >= total) return;
    int gr = i >> 4, o = i & 15;
    float s = bl2[o];
    for (int j = 0; j < 32; ++j) s += t[gr * 32 + j] * Wl2[j * OUT_C + o];
    out[i] = s;
}

// ---------------- launcher ----------------

extern "C" void kernel_launch(void* const* d_in, const int* in_sizes, int n_in,
                              void* d_out, int out_size, void* d_ws, size_t ws_size,
                              hipStream_t stream) {
    (void)in_sizes; (void)n_in; (void)out_size; (void)ws_size;
    const float* x     = (const float*)d_in[0];
    const int*   ei    = (const int*)d_in[1];
    const int*   src   = ei;
    const int*   dst   = ei + N_EDGES;
    const float* ew    = (const float*)d_in[2];
    const int*   batch = (const int*)d_in[3];
    const float* bn_g  = (const float*)d_in[4];
    const float* bn_b  = (const float*)d_in[5];
    const float* bn_m  = (const float*)d_in[6];
    const float* bn_v  = (const float*)d_in[7];
    const float* Wm[3]   = { (const float*)d_in[8],  (const float*)d_in[14], (const float*)d_in[20] };
    const float* attS[3] = { (const float*)d_in[9],  (const float*)d_in[15], (const float*)d_in[21] };
    const float* attD[3] = { (const float*)d_in[10], (const float*)d_in[16], (const float*)d_in[22] };
    const float* Wep[3]  = { (const float*)d_in[11], (const float*)d_in[17], (const float*)d_in[23] };
    const float* attE[3] = { (const float*)d_in[12], (const float*)d_in[18], (const float*)d_in[24] };
    const float* bp[3]   = { (const float*)d_in[13], (const float*)d_in[19], (const float*)d_in[25] };
    const float* Wl1 = (const float*)d_in[26];
    const float* bl1 = (const float*)d_in[27];
    const float* Wl2 = (const float*)d_in[28];
    const float* bl2 = (const float*)d_in[29];
    float* out = (float*)d_out;

    // workspace partition (floats)
    const size_t NB = (size_t)N_NODES * H_DIM;           // 6.4M
    float* ws     = (float*)d_ws;
    float* bufX   = ws;                                  // layer input features
    float* bufL   = ws + NB;                             // h = X @ W
    float* bufC   = ws + 2 * NB;                         // scatter accumulator
    float* deg    = ws + 3 * NB;                         // N
    float* eamean = deg + N_NODES;                       // N (sum, then mean in place)
    float* aS     = eamean + N_NODES;                    // N*8
    float* aD     = aS + 8 * N_NODES;                    // N*8
    float* amax   = aD + 8 * N_NODES;                    // N*8
    float* denom  = amax + 8 * N_NODES;                  // N*8
    float* ce     = denom + 8 * N_NODES;                 // 8
    float* gpool  = ce + 8;                              // G*128
    float* tmlp   = gpool + (size_t)G_GR * H_DIM;        // G*32
    __bf16* Wpk   = (__bf16*)(tmlp + (size_t)G_GR * 32); // 16384 bf16 (32B-aligned offset)

    const int BS = 256;
    const int EF = N_EDGES + N_NODES;                    // edges incl. self-loops
    const float NEG_INF = -INFINITY;

    // BatchNorm
    bn_kernel<<<CDIV(N_NODES * H_DIM, BS), BS, 0, stream>>>(x, bn_g, bn_b, bn_m, bn_v,
                                                            bufX, N_NODES * H_DIM);
    // degree + mean incoming edge attr (graph-constant across layers)
    fill_kernel<<<CDIV(2 * N_NODES, BS), BS, 0, stream>>>(deg, 0.0f, 2 * N_NODES);
    deg_easum_kernel<<<CDIV(N_EDGES, BS), BS, 0, stream>>>(dst, ew, deg, eamean, N_EDGES);
    eamean_kernel<<<CDIV(N_NODES, BS), BS, 0, stream>>>(deg, eamean, N_NODES);

    for (int l = 0; l < 3; ++l) {
        ce_kernel<<<1, 32, 0, stream>>>(Wep[l], attE[l], ce);
        packW_kernel<<<CDIV(16384, BS), BS, 0, stream>>>(Wm[l], Wpk);
        gemm_wmma_kernel<<<CDIV(N_NODES / 16, 4), 128, 0, stream>>>(bufX, Wpk, bufL);
        node_coef_kernel<<<CDIV(N_NODES * HEADS, BS), BS, 0, stream>>>(bufL, attS[l], attD[l],
                                                                       aS, aD, N_NODES * HEADS);
        fill_kernel<<<CDIV(8 * N_NODES, BS), BS, 0, stream>>>(amax, NEG_INF, 8 * N_NODES);
        fill_kernel<<<CDIV(8 * N_NODES, BS), BS, 0, stream>>>(denom, 0.0f, 8 * N_NODES);
        fill_kernel<<<CDIV(N_NODES * H_DIM, BS), BS, 0, stream>>>(bufC, 0.0f, N_NODES * H_DIM);
        edge_max_kernel<<<CDIV(EF * HEADS, BS), BS, 0, stream>>>(src, dst, ew, eamean, aS, aD,
                                                                 ce, amax, EF * HEADS);
        edge_expsum_kernel<<<CDIV(EF * HEADS, BS), BS, 0, stream>>>(src, dst, ew, eamean, aS, aD,
                                                                    ce, amax, denom, EF * HEADS);
        edge_scatter_kernel<<<CDIV(EF * 32, BS), BS, 0, stream>>>(src, dst, ew, eamean, aS, aD,
                                                                  ce, amax, denom, bufL, bufC,
                                                                  EF * 32);
        bias_act_kernel<<<CDIV(N_NODES * H_DIM, BS), BS, 0, stream>>>(bufC, bp[l], bufX,
                                                                      N_NODES * H_DIM, l < 2 ? 1 : 0);
    }

    // global max pool over graphs + MLP head
    fill_kernel<<<CDIV(G_GR * H_DIM, BS), BS, 0, stream>>>(gpool, NEG_INF, G_GR * H_DIM);
    pool_kernel<<<CDIV(N_NODES * H_DIM, BS), BS, 0, stream>>>(bufX, batch, gpool, N_NODES * H_DIM);
    mlp1_kernel<<<CDIV(G_GR * 32, BS), BS, 0, stream>>>(gpool, Wl1, bl1, tmlp, G_GR * 32);
    mlp2_kernel<<<CDIV(G_GR * OUT_C, BS), BS, 0, stream>>>(tmlp, Wl2, bl2, out, G_GR * OUT_C);
}
